// MultiHeadAttention_86741159510253
// MI455X (gfx1250) — matrix operand order repaired and checked by reading, not compiled, not run on hardware
//
#include <hip/hip_runtime.h>
#include <hip/hip_bf16.h>

// MQA attention for MI455X (gfx1250, wave32, WMMA), bf16 compute / f32 accumulate.
// v3b: big GEMMs use cooperative LDS staging, double-buffered, fed by gfx1250
// async global->LDS copies (ASYNCcnt) when available; operand reads are
// ds_load_b128 in exact WMMA register layout. Attention softmax de-VALU'd
// (affine ALiBi, no int->float cvts in the key loop).

#define T_SEQ 2048
#define EMB   2048
#define NH    32
#define HD    64

typedef __attribute__((ext_vector_type(16))) __bf16 v16bf;
typedef __attribute__((ext_vector_type(8)))  float  v8f;
typedef __attribute__((ext_vector_type(4)))  int    v4i;

union BfVec {
    unsigned short s[16];
    uint4          q[2];
    v16bf          v;
};

#if defined(__AMDGCN__) && __has_builtin(__builtin_amdgcn_global_load_async_to_lds_b128)
#define USE_ASYNC_LDS 1
#else
#define USE_ASYNC_LDS 0
#endif

// round-to-nearest-even f32 -> bf16 (bit pattern)
__device__ __forceinline__ unsigned short f2bf(float f) {
    union { float f; unsigned u; } c;
    c.f = f;
    unsigned u = c.u + 0x7FFFu + ((c.u >> 16) & 1u);
    return (unsigned short)(u >> 16);
}

__device__ __forceinline__ v8f wmma_bf16(const BfVec& a, const BfVec& b, v8f c) {
    return __builtin_amdgcn_wmma_f32_16x16x32_bf16(false, a.v, false, b.v,
                                                   (short)0, c, false, false);
}

// A operand (16x32 bf16): lane = 16g+m holds row m, K = {8g..8g+7, 16+8g..+7}.
__device__ __forceinline__ void fillA(BfVec& a, const unsigned short* p) {
    a.q[0] = *(const uint4*)(p);
    a.q[1] = *(const uint4*)(p + 16);
}

// B operand (32x16 bf16): lane = 16g+n holds col n, K = {8g..8g+7, 16+8g..+7}.
__device__ __forceinline__ void fillB(BfVec& b, const unsigned short* p) {
    b.q[0] = *(const uint4*)(p);
    b.q[1] = *(const uint4*)(p + 16);
}

// 16B global -> LDS copy: async (ASYNCcnt-tracked) or sync fallback.
__device__ __forceinline__ void copy16B(const unsigned short* gp, unsigned short* lp) {
#if USE_ASYNC_LDS
    __builtin_amdgcn_global_load_async_to_lds_b128(
        (__attribute__((address_space(1))) v4i*)gp,
        (__attribute__((address_space(3))) v4i*)lp, 0, 0);
#else
    *(uint4*)lp = *(const uint4*)gp;
#endif
}

__device__ __forceinline__ void wait_copies() {
#if USE_ASYNC_LDS
    asm volatile("s_wait_asynccnt 0" ::: "memory");
#endif
}

// elementwise f32 -> bf16, 8 elements per thread
__global__ __launch_bounds__(256) void cvt_f32_bf16(
    const float* __restrict__ in, unsigned short* __restrict__ out, int n)
{
    int i = (blockIdx.x * 256 + threadIdx.x) * 8;
    if (i >= n) return;
    float4 f0 = *(const float4*)(in + i);
    float4 f1 = *(const float4*)(in + i + 4);
    union { unsigned short s[8]; uint4 q; } o;
    o.s[0] = f2bf(f0.x); o.s[1] = f2bf(f0.y);
    o.s[2] = f2bf(f0.z); o.s[3] = f2bf(f0.w);
    o.s[4] = f2bf(f1.x); o.s[5] = f2bf(f1.y);
    o.s[6] = f2bf(f1.z); o.s[7] = f2bf(f1.w);
    *(uint4*)(out + i) = o.q;
}

// ---------------------------------------------------------------------------
// Staged GEMM: C = A(MxK) * B(NxK)^T, bf16. Block = 256 threads = 8 waves,
// block tile 64(M) x 128(N), waves 2x4, 32x32 per wave, k-step 32,
// double-buffered LDS fed by async copies. Requires M%64==0, N%128==0, K%32==0.
// ---------------------------------------------------------------------------
template <int BF16OUT>
__global__ __launch_bounds__(256) void gemm_staged(
    const unsigned short* __restrict__ A, const unsigned short* __restrict__ B,
    void* __restrict__ Cv, int M, int N, int K, int lda, int ldb, int ldc)
{
    __shared__ __align__(16) unsigned short As[2][64][40];   // 80B pitch
    __shared__ __align__(16) unsigned short Bs[2][128][40];

    const int t    = threadIdx.x;
    const int lane = t & 31;
    const int w    = t >> 5;
    const int wm   = w >> 2;          // 0..1
    const int wn   = w & 3;           // 0..3
    const int g    = lane >> 4;
    const int ln   = lane & 15;

    const int nbN = N >> 7;
    const int mt  = blockIdx.x / nbN;
    const int nt  = blockIdx.x % nbN;
    const int mBase = mt * 64;
    const int nBase = nt * 128;

    const int row = t >> 2;           // 0..63
    const int ch  = t & 3;            // 0..3 (16B chunks of a 32-elem row)

    v8f acc00 = {}, acc01 = {}, acc10 = {}, acc11 = {};

    auto stage = [&](int kb, int buf) {
        copy16B(A + (size_t)(mBase + row) * lda + kb + ch * 8,       &As[buf][row][ch * 8]);
        copy16B(B + (size_t)(nBase + row) * ldb + kb + ch * 8,       &Bs[buf][row][ch * 8]);
        copy16B(B + (size_t)(nBase + 64 + row) * ldb + kb + ch * 8,  &Bs[buf][64 + row][ch * 8]);
    };

    const int nk = K >> 5;
    stage(0, 0);
    for (int k = 0; k < nk; ++k) {
        wait_copies();            // my batch-k copies done (ASYNCcnt -> 0)
        __syncthreads();          // everyone's batch-k data visible; buf k-1 free
        if (k + 1 < nk) stage((k + 1) << 5, (k + 1) & 1);

        const unsigned short (*Ab)[40] = As[k & 1];
        const unsigned short (*Bb)[40] = Bs[k & 1];
        BfVec a0, a1, b0, b1;
        fillA(a0, &Ab[wm * 32 + ln][g * 8]);
        fillA(a1, &Ab[wm * 32 + 16 + ln][g * 8]);
        fillB(b0, &Bb[wn * 32 + ln][g * 8]);
        fillB(b1, &Bb[wn * 32 + 16 + ln][g * 8]);
        acc00 = wmma_bf16(a0, b0, acc00);
        acc01 = wmma_bf16(a0, b1, acc01);
        acc10 = wmma_bf16(a1, b0, acc10);
        acc11 = wmma_bf16(a1, b1, acc11);
    }

#pragma unroll
    for (int i = 0; i < 2; ++i) {
#pragma unroll
        for (int j = 0; j < 2; ++j) {
            const v8f acc = (i == 0) ? (j == 0 ? acc00 : acc01)
                                     : (j == 0 ? acc10 : acc11);
#pragma unroll
            for (int r = 0; r < 8; ++r) {
                const int m = mBase + wm * 32 + i * 16 + r + 8 * g;
                const int n = nBase + wn * 32 + j * 16 + ln;
                if (BF16OUT) ((unsigned short*)Cv)[(size_t)m * ldc + n] = f2bf(acc[r]);
                else         ((float*)Cv)[(size_t)m * ldc + n]          = acc[r];
            }
        }
    }
}

// Small direct GEMM (for the 64-wide K/V projections). 32x32 tile per wave.
template <int TRANSC>
__global__ __launch_bounds__(256) void gemm_small(
    const unsigned short* __restrict__ A, const unsigned short* __restrict__ B,
    unsigned short* __restrict__ C, int M, int N, int K, int lda, int ldb, int ldc)
{
    const int lane = threadIdx.x & 31;
    const int w    = threadIdx.x >> 5;
    const int tile = blockIdx.x * 8 + w;
    const int ntN  = N >> 5;
    if (tile >= (M >> 5) * ntN) return;      // whole-wave guard

    const int mt = tile / ntN;
    const int nt = tile % ntN;
    const int g  = lane >> 4;
    const int ln = lane & 15;

    const unsigned short* arow0 = A + (size_t)(mt * 32 + ln) * lda;
    const unsigned short* arow1 = arow0 + (size_t)16 * lda;
    const unsigned short* brow0 = B + (size_t)(nt * 32 + ln) * ldb;
    const unsigned short* brow1 = brow0 + (size_t)16 * ldb;

    v8f acc00 = {}, acc01 = {}, acc10 = {}, acc11 = {};
    for (int kb = 0; kb < K; kb += 32) {
        BfVec a0, a1, b0, b1;
        fillA(a0, arow0 + kb + g * 8);
        fillA(a1, arow1 + kb + g * 8);
        fillB(b0, brow0 + kb + g * 8);
        fillB(b1, brow1 + kb + g * 8);
        acc00 = wmma_bf16(a0, b0, acc00);
        acc01 = wmma_bf16(a0, b1, acc01);
        acc10 = wmma_bf16(a1, b0, acc10);
        acc11 = wmma_bf16(a1, b1, acc11);
    }
#pragma unroll
    for (int i = 0; i < 2; ++i) {
#pragma unroll
        for (int j = 0; j < 2; ++j) {
            const v8f acc = (i == 0) ? (j == 0 ? acc00 : acc01)
                                     : (j == 0 ? acc10 : acc11);
#pragma unroll
            for (int r = 0; r < 8; ++r) {
                const int m = mt * 32 + i * 16 + r + 8 * g;
                const int n = nt * 32 + j * 16 + ln;
                const size_t idx = TRANSC ? (size_t)n * ldc + m : (size_t)m * ldc + n;
                C[idx] = f2bf(acc[r]);
            }
        }
    }
}

// ---------------------------------------------------------------------------
// Flash attention: one wave = 16 query rows of one head. All inputs bf16.
// ---------------------------------------------------------------------------
__global__ __launch_bounds__(256) void attn_wmma_kernel(
    const unsigned short* __restrict__ Q, const unsigned short* __restrict__ Km,
    const unsigned short* __restrict__ Vt, unsigned short* __restrict__ O)
{
    __shared__ __align__(16) unsigned short Plds[8][16][40];

    const int lane  = threadIdx.x & 31;
    const int wslot = threadIdx.x >> 5;
    const int wid   = blockIdx.x * 8 + wslot;
    const int h     = wid >> 7;
    const int q0    = (wid & 127) * 16;
    const int g     = lane >> 4;
    const int ln    = lane & 15;

    const float slope = exp2f(-0.25f * (float)(h + 1));
    const float scale = 0.125f;

    unsigned short (*P)[40] = Plds[wslot];

    const unsigned short* qrow = Q + (size_t)(q0 + ln) * EMB + h * HD;
    BfVec qa0, qa1;
    fillA(qa0, qrow + g * 8);
    fillA(qa1, qrow + 32 + g * 8);

    float rowmax[8], rowsum[8], slopeq[8];
    v8f acc[4];
#pragma unroll
    for (int r = 0; r < 8; ++r) {
        rowmax[r] = -1e30f;
        rowsum[r] = 0.0f;
        slopeq[r] = slope * (float)(q0 + r + 8 * g);   // slope * qpos (hoisted)
    }
#pragma unroll
    for (int c = 0; c < 4; ++c) acc[c] = (v8f){};

    float colb = slope * (float)ln;                     // slope * (kb + ln), kb=0
    const float colStep16 = 16.0f * slope;
    const float colStep32 = 32.0f * slope;

    const int kmax = q0 + 15;
    for (int kb = 0; kb <= kmax; kb += 32) {
        asm volatile("" ::: "memory");

        v8f s0 = {}, s1 = {};
        {
            BfVec bk;
            const unsigned short* kp0 = Km + (size_t)(kb + ln) * HD + g * 8;
            const unsigned short* kp1 = Km + (size_t)(kb + 16 + ln) * HD + g * 8;
            fillB(bk, kp0);       s0 = wmma_bf16(qa0, bk, s0);
            fillB(bk, kp0 + 32);  s0 = wmma_bf16(qa1, bk, s0);
            fillB(bk, kp1);       s1 = wmma_bf16(qa0, bk, s1);
            fillB(bk, kp1 + 32);  s1 = wmma_bf16(qa1, bk, s1);
        }

#pragma unroll
        for (int r = 0; r < 8; ++r) {
            const int m    = r + 8 * g;
            const int qpos = q0 + m;
            const int k0   = kb + ln;
            const int k1   = k0 + 16;
            // alibi bias = colb - slopeq[r] (affine; no int->float cvt here)
            float sc0 = (k0 > qpos) ? -10000.0f : fmaf(s0[r], scale, colb - slopeq[r]);
            float sc1 = (k1 > qpos) ? -10000.0f
                                    : fmaf(s1[r], scale, colb + colStep16 - slopeq[r]);
            float mx = fmaxf(sc0, sc1);
            mx = fmaxf(mx, __shfl_xor(mx, 1));
            mx = fmaxf(mx, __shfl_xor(mx, 2));
            mx = fmaxf(mx, __shfl_xor(mx, 4));
            mx = fmaxf(mx, __shfl_xor(mx, 8));
            const float nm   = fmaxf(rowmax[r], mx);
            const float corr = __expf(rowmax[r] - nm);
            rowmax[r] = nm;
            const float e0 = __expf(sc0 - nm);
            const float e1 = __expf(sc1 - nm);
            float ls = e0 + e1;
            ls += __shfl_xor(ls, 1);
            ls += __shfl_xor(ls, 2);
            ls += __shfl_xor(ls, 4);
            ls += __shfl_xor(ls, 8);
            rowsum[r] = rowsum[r] * corr + ls;
            acc[0][r] *= corr; acc[1][r] *= corr;
            acc[2][r] *= corr; acc[3][r] *= corr;
            P[m][ln]      = f2bf(e0);
            P[m][16 + ln] = f2bf(e1);
        }
        colb += colStep32;

        asm volatile("s_wait_dscnt 0" ::: "memory");   // same-wave LDS RAW

        BfVec pa;
        pa.q[0] = *(const uint4*)&P[ln][g * 8];
        pa.q[1] = *(const uint4*)&P[ln][16 + g * 8];

#pragma unroll
        for (int c = 0; c < 4; ++c) {
            BfVec bv;
            fillB(bv, Vt + (size_t)(c * 16 + ln) * T_SEQ + kb + g * 8);
            acc[c] = wmma_bf16(pa, bv, acc[c]);
        }
    }

#pragma unroll
    for (int r = 0; r < 8; ++r) {
        const int m = r + 8 * g;
        const float inv = 1.0f / rowsum[r];
#pragma unroll
        for (int c = 0; c < 4; ++c)
            O[(size_t)(q0 + m) * EMB + h * HD + c * 16 + ln] = f2bf(acc[c][r] * inv);
    }
}

extern "C" void kernel_launch(void* const* d_in, const int* in_sizes, int n_in,
                              void* d_out, int out_size, void* d_ws, size_t ws_size,
                              hipStream_t stream) {
    const float* x  = (const float*)d_in[0];
    const float* Wq = (const float*)d_in[1];
    const float* Wk = (const float*)d_in[2];
    const float* Wv = (const float*)d_in[3];
    const float* Wo = (const float*)d_in[4];
    // d_in[5] = causal mask: computed analytically in-kernel.

    const size_t NE = (size_t)T_SEQ * EMB;   // 4M
    const size_t ND = (size_t)T_SEQ * HD;    // 128K

    unsigned short* xb  = (unsigned short*)d_ws;
    unsigned short* Wqb = xb  + NE;
    unsigned short* Wob = Wqb + NE;
    unsigned short* Wkb = Wob + NE;
    unsigned short* Wvb = Wkb + ND;
    unsigned short* Qb  = Wvb + ND;
    unsigned short* Kb  = Qb  + NE;
    unsigned short* Vtb = Kb  + ND;
    unsigned short* AOb = Vtb + ND;
    float* out = (float*)d_out;

    cvt_f32_bf16<<<(int)(NE / 2048), 256, 0, stream>>>(x,  xb,  (int)NE);
    cvt_f32_bf16<<<(int)(NE / 2048), 256, 0, stream>>>(Wq, Wqb, (int)NE);
    cvt_f32_bf16<<<(int)(NE / 2048), 256, 0, stream>>>(Wo, Wob, (int)NE);
    cvt_f32_bf16<<<(int)(ND / 2048), 256, 0, stream>>>(Wk, Wkb, (int)ND);
    cvt_f32_bf16<<<(int)(ND / 2048), 256, 0, stream>>>(Wv, Wvb, (int)ND);

    // Qb = x @ Wq^T  (staged: 32 x 16 = 512 blocks)
    gemm_staged<1><<<512, 256, 0, stream>>>(xb, Wqb, Qb,
        T_SEQ, EMB, EMB, EMB, EMB, EMB);
    // Kb = x @ Wk^T  [T, 64]
    gemm_small<0><<<16, 256, 0, stream>>>(xb, Wkb, Kb,
        T_SEQ, HD, EMB, EMB, EMB, HD);
    // Vtb = (x @ Wv^T)^T  [64, T]
    gemm_small<1><<<16, 256, 0, stream>>>(xb, Wvb, Vtb,
        T_SEQ, HD, EMB, EMB, EMB, T_SEQ);
    // attention: 4096 waves
    attn_wmma_kernel<<<512, 256, 0, stream>>>(Qb, Kb, Vtb, AOb);
    // out = AO @ Wo^T (f32 out, staged)
    gemm_staged<0><<<512, 256, 0, stream>>>(AOb, Wob, out,
        T_SEQ, EMB, EMB, EMB, EMB, EMB);
}
